// SchNetAdapt_38311108281152
// MI455X (gfx1250) — matrix operand order repaired and checked by reading, not compiled, not run on hardware
//
#include <hip/hip_runtime.h>
#include <math.h>

typedef _Float16 half_t;
typedef __attribute__((ext_vector_type(16))) _Float16 v16h;
typedef __attribute__((ext_vector_type(8)))  _Float16 v8h;
typedef __attribute__((ext_vector_type(8)))  float    v8f;

#define N_NODES 20000
#define N_EDGES 640000
#define HID 128
#define NBF 3
#define NGSF 47
#define CUTF 10.0f
#define NLAYER 3
#define NBATCH 16
#define LOG2F_C 0.6931471805599453f
#define PI_F 3.14159265358979f

__device__ __forceinline__ float sspf(float x) {
  float sp = (x > 20.f) ? x : __logf(1.f + __expf(x));
  return sp - LOG2F_C;
}

__device__ __forceinline__ v8f wmma32(v16h a, v16h b, v8f c) {
  // D = A(16x32 f16) * B(32x16 f16) + C(16x16 f32)
  return __builtin_amdgcn_wmma_f32_16x16x32_f16(false, a, false, b, (short)0, c,
                                                false, false);
}

// Build a 16x32 f16 A-fragment from a per-lane base pointer (already offset to
// this lane's row and half-select). Two contiguous 16B loads.
// ISA layout: lanes 0-15 hold K {k0..k0+7, k0+16..k0+23}; lanes 16-31 same +8.
__device__ __forceinline__ v16h frag_a_at(const half_t* p) {
  v8h x0 = *(const v8h*)p;
  v8h x1 = *(const v8h*)(p + 16);
  v16h r;
#pragma unroll
  for (int i = 0; i < 8; ++i) { r[i] = x0[i]; r[i + 8] = x1[i]; }
  return r;
}

// Async copy 16B/lane chunks from global into LDS (CDNA5 ASYNCcnt path).
__device__ __forceinline__ void async_copy16(uint32_t lds_off,
                                             const void* gptr) {
  uint64_t ga = (uint64_t)(uintptr_t)gptr;
  asm volatile("global_load_async_to_lds_b128 %0, %1, off"
               :
               : "v"(lds_off), "v"(ga)
               : "memory");
}

__device__ __forceinline__ void wait_async0() {
  asm volatile("s_wait_asynccnt 0x0" ::: "memory");
}

// ---------- weight pre-pack into per-lane B-fragment layout ----------
// packed[((nt*nkt + kt)*32 + lane)*16 + h] = W[k, nt*16 + (lane&15)]
// with k = kt*32 + ((lane&16)?8:0) + (h&7) + ((h&8)?16:0)   (zero-fill for k >= K)
__global__ void pack_b_kernel(const float* __restrict__ W,
                              half_t* __restrict__ out,
                              int K, int N, int nkt) {
  int idx = blockIdx.x * blockDim.x + threadIdx.x;
  int total = (N >> 4) * nkt * 512;
  if (idx >= total) return;
  int h    = idx & 15;
  int lane = (idx >> 4) & 31;
  int kt   = (idx >> 9) % nkt;
  int nt   = idx / (nkt << 9);
  int k = kt * 32 + ((lane & 16) ? 8 : 0) + (h & 7) + ((h & 8) ? 16 : 0);
  int n = nt * 16 + (lane & 15);
  out[idx] = (k < K) ? (half_t)W[(size_t)k * N + n] : (half_t)0.f;
}

// ---------- misc small kernels ----------
__global__ void zero_kernel(float* p, int n) {
  int i = blockIdx.x * blockDim.x + threadIdx.x;
  if (i < n) p[i] = 0.f;
}

__global__ void f32to16_kernel(const float* __restrict__ a,
                               half_t* __restrict__ b, int n) {
  int i = blockIdx.x * blockDim.x + threadIdx.x;
  if (i < n) b[i] = (half_t)a[i];
}

__global__ void embed_kernel(const int* __restrict__ z,
                             const float* __restrict__ emb,
                             float* __restrict__ h, half_t* __restrict__ h16) {
  int i = blockIdx.x * blockDim.x + threadIdx.x;
  if (i >= N_NODES * HID) return;
  int n = i >> 7, c = i & 127;
  float v = emb[(size_t)z[n] * HID + c];
  h[i] = v;
  h16[i] = (half_t)v;
}

// ---------- generic WMMA GEMM (K fixed at 128): out = act(A @ Bp + bias) ----
// grid.x = M/16 (one 16-row stripe per block); wave w handles column tile w.
// flags bit0: ShiftedSoftplus. bit1: residual add (read o32 first).
__global__ void __launch_bounds__(256)
gemm16_kernel(const half_t* __restrict__ A, const half_t* __restrict__ Bp,
              const float* __restrict__ bias, float* __restrict__ o32,
              half_t* __restrict__ o16, int N, int flags) {
  const int lane = threadIdx.x & 31;
  const int w    = threadIdx.x >> 5;
  if (w * 16 >= N) return;  // wave-uniform: EXEC stays all-ones for WMMA
  const int mb   = blockIdx.x << 4;
  const int mrow = mb + (lane & 15);
  // per-lane base pointers; all kt steps become immediate offsets
  const half_t* Ap = A + (size_t)mrow * 128 + ((lane & 16) ? 8 : 0);
  const half_t* Bw = Bp + ((size_t)w * 128 + lane) * 16;  // w*nkt(4)*32+lane
  v8f acc = {};
#pragma unroll
  for (int kt = 0; kt < 4; ++kt) {
    v16h a = frag_a_at(Ap + kt * 32);
    v16h b = *(const v16h*)(Bw + kt * 512);
    acc = wmma32(a, b, acc);
  }
  const int   c  = (w << 4) + (lane & 15);
  const float bb = bias ? bias[c] : 0.f;
  const int   mo = (lane & 16) ? 8 : 0;
#pragma unroll
  for (int v = 0; v < 8; ++v) {
    int m = mb + mo + v;
    float val = acc[v] + bb;
    if (flags & 1) val = sspf(val);
    size_t idx = (size_t)m * N + c;
    if (flags & 2) val += o32[idx];
    if (o32) o32[idx] = val;
    if (o16) o16[idx] = (half_t)val;
  }
}

// ---------- fused per-layer edge pipeline ----------
// Block = 256 threads (8 waves) handles 128 edges. Dynamic LDS layout:
//   [w1S 16KB][w2S 32KB][eaS 16KB][h1S 32KB][Cs][rowS][colS]
// Weights are async-copied global->LDS, overlapped with Gaussian smearing.
#define SM_W1 0
#define SM_W2 16384
#define SM_EA 49152
#define SM_H1 65536
#define SM_CS 98304
#define SM_ROW 98816
#define SM_COL 99328
#define SM_TOTAL 99840

__global__ void __launch_bounds__(256)
edge_layer_kernel(const int* __restrict__ ei, const float* __restrict__ pos,
                  const float* __restrict__ eattr,
                  const half_t* __restrict__ w1p, const float* __restrict__ b1,
                  const half_t* __restrict__ w2p, const float* __restrict__ b2,
                  const float* __restrict__ xh, float* __restrict__ agg) {
  extern __shared__ __align__(16) char smem[];
  half_t* w1S  = (half_t*)(smem + SM_W1);
  half_t* w2S  = (half_t*)(smem + SM_W2);
  half_t* eaS  = (half_t*)(smem + SM_EA);
  half_t* h1S  = (half_t*)(smem + SM_H1);
  float*  Cs   = (float*)(smem + SM_CS);
  int*    rowS = (int*)(smem + SM_ROW);
  int*    colS = (int*)(smem + SM_COL);

  const int t    = threadIdx.x;
  const int lane = t & 31;
  const int wv   = t >> 5;
  const int ebase = blockIdx.x * 128;

  // kick off async weight staging (12 x 16B per lane), hidden behind stage 0
  const uint32_t w1base = (uint32_t)(uintptr_t)w1S;
  const uint32_t w2base = (uint32_t)(uintptr_t)w2S;
#pragma unroll
  for (int i = 0; i < 4; ++i) {  // 16KB: 1024 chunks of 16B
    int ck = t + i * 256;
    async_copy16(w1base + ck * 16, (const char*)w1p + (size_t)ck * 16);
  }
#pragma unroll
  for (int i = 0; i < 8; ++i) {  // 32KB: 2048 chunks of 16B
    int ck = t + i * 256;
    async_copy16(w2base + ck * 16, (const char*)w2p + (size_t)ck * 16);
  }

  // stage 0: edge features (Gaussian smearing + cutoff) -> LDS
  if (t < 128) {
    int e  = ebase + t;
    int r  = ei[e];
    int cc = ei[N_EDGES + e];
    rowS[t] = r;
    colS[t] = cc;
    float dx = pos[r * 3 + 0] - pos[cc * 3 + 0];
    float dy = pos[r * 3 + 1] - pos[cc * 3 + 1];
    float dz = pos[r * 3 + 2] - pos[cc * 3 + 2];
    float d  = sqrtf(dx * dx + dy * dy + dz * dz);
    Cs[t] = 0.5f * (__cosf(d * (PI_F / CUTF)) + 1.f);
    eaS[t * 64 + 0] = (half_t)eattr[(size_t)e * 3 + 0];
    eaS[t * 64 + 1] = (half_t)eattr[(size_t)e * 3 + 1];
    eaS[t * 64 + 2] = (half_t)eattr[(size_t)e * 3 + 2];
    const float step  = CUTF / (float)(NGSF - 1);
    const float coeff = -0.5f / (step * step);
#pragma unroll
    for (int g = 0; g < NGSF; ++g) {
      float diff = d - step * (float)g;
      eaS[t * 64 + NBF + g] = (half_t)__expf(coeff * diff * diff);
    }
#pragma unroll
    for (int p = NBF + NGSF; p < 64; ++p) eaS[t * 64 + p] = (half_t)0.f;
  }
  wait_async0();     // this wave's LDS weight chunks have landed
  __syncthreads();   // all waves' chunks + all edge features visible

  const int mt   = wv;  // each wave owns one 16-edge stripe
  const int mo   = (lane & 16) ? 8 : 0;
  const int nidx = lane & 15;
  const half_t* b1lane = w1S + (size_t)lane * 16;
  const half_t* b2lane = w2S + (size_t)lane * 16;

  // stage 1: hidden1 = ssp(ea @ W1 + b1), K = 64. A-frags hoisted (reused x8).
  {
    const half_t* ap = eaS + (size_t)(mt * 16 + nidx) * 64 + ((lane & 16) ? 8 : 0);
    v16h a0 = frag_a_at(ap);
    v16h a1 = frag_a_at(ap + 32);
#pragma unroll
    for (int nt = 0; nt < 8; ++nt) {
      v8f acc = {};
      acc = wmma32(a0, *(const v16h*)(b1lane + (nt * 2 + 0) * 512), acc);
      acc = wmma32(a1, *(const v16h*)(b1lane + (nt * 2 + 1) * 512), acc);
      int c = nt * 16 + nidx;
      float bb = b1[c];
#pragma unroll
      for (int v = 0; v < 8; ++v)
        h1S[(mt * 16 + mo + v) * 128 + c] = (half_t)sspf(acc[v] + bb);
    }
  }
  __syncthreads();

  // stage 2: W = (hidden1 @ W2 + b2)*C, K = 128, fused gather-mul-scatter.
  {
    const half_t* ap = h1S + (size_t)(mt * 16 + nidx) * 128 + ((lane & 16) ? 8 : 0);
    v16h a0 = frag_a_at(ap);
    v16h a1 = frag_a_at(ap + 32);
    v16h a2 = frag_a_at(ap + 64);
    v16h a3 = frag_a_at(ap + 96);
#pragma unroll
    for (int nt = 0; nt < 8; ++nt) {
      v8f acc = {};
      acc = wmma32(a0, *(const v16h*)(b2lane + (nt * 4 + 0) * 512), acc);
      acc = wmma32(a1, *(const v16h*)(b2lane + (nt * 4 + 1) * 512), acc);
      acc = wmma32(a2, *(const v16h*)(b2lane + (nt * 4 + 2) * 512), acc);
      acc = wmma32(a3, *(const v16h*)(b2lane + (nt * 4 + 3) * 512), acc);
      int c = nt * 16 + nidx;
      float bb = b2[c];
#pragma unroll
      for (int v = 0; v < 8; ++v) {
        int   el  = mt * 16 + mo + v;
        float Wv  = (acc[v] + bb) * Cs[el];
        float msg = Wv * xh[(size_t)rowS[el] * HID + c];
        atomicAdd(&agg[(size_t)colS[el] * HID + c], msg);
      }
    }
  }
}

// ---------- final readout: per-node matvec (H/2 -> 1) + batch scatter -------
__global__ void readout_kernel(const half_t* __restrict__ o1,
                               const float* __restrict__ w2,
                               const float* __restrict__ b2,
                               const int* __restrict__ batch,
                               float* __restrict__ out) {
  int n = blockIdx.x * blockDim.x + threadIdx.x;
  if (n >= N_NODES) return;
  float acc = b2[0];
#pragma unroll
  for (int j = 0; j < 64; ++j) acc += (float)o1[(size_t)n * 64 + j] * w2[j];
  atomicAdd(&out[batch[n]], acc);
}

extern "C" void kernel_launch(void* const* d_in, const int* in_sizes, int n_in,
                              void* d_out, int out_size, void* d_ws,
                              size_t ws_size, hipStream_t stream) {
  const int*   z     = (const int*)d_in[0];
  const float* pos   = (const float*)d_in[1];
  const int*   batch = (const int*)d_in[2];
  const int*   eidx  = (const int*)d_in[3];
  const float* eattr = (const float*)d_in[4];
  const float* emb   = (const float*)d_in[5];
  const float* w1    = (const float*)d_in[6];
  const float* b1    = (const float*)d_in[7];
  const float* w2    = (const float*)d_in[8];
  const float* b2    = (const float*)d_in[9];
  const float* cf1   = (const float*)d_in[10];
  const float* cf2   = (const float*)d_in[11];
  const float* cf2b  = (const float*)d_in[12];
  const float* linw  = (const float*)d_in[13];
  const float* linb  = (const float*)d_in[14];
  const float* o1w   = (const float*)d_in[15];
  const float* o1b   = (const float*)d_in[16];
  const float* o2w   = (const float*)d_in[17];
  const float* o2b   = (const float*)d_in[18];
  float* out = (float*)d_out;

  // ---- workspace carving (~49 MB) ----
  char* ws = (char*)d_ws;
  size_t off = 0;
  auto carve = [&](size_t bytes) -> void* {
    void* p = ws + off;
    off = (off + bytes + 255) & ~(size_t)255;
    return p;
  };
  float*  hbuf  = (float*)carve((size_t)N_NODES * HID * 4);
  half_t* h16   = (half_t*)carve((size_t)N_NODES * HID * 2);
  float*  xh    = (float*)carve((size_t)N_NODES * HID * 4);
  float*  agg   = (float*)carve((size_t)N_NODES * HID * 4);
  half_t* agg16 = (half_t*)carve((size_t)N_NODES * HID * 2);
  half_t* t16   = (half_t*)carve((size_t)N_NODES * HID * 2);
  half_t* o1buf = (half_t*)carve((size_t)N_NODES * 64 * 2);
  half_t *w1p[NLAYER], *w2p[NLAYER], *cf1p[NLAYER], *cf2p[NLAYER], *linp[NLAYER];
  for (int l = 0; l < NLAYER; ++l) w1p[l]  = (half_t*)carve(8 * 2 * 512 * 2);
  for (int l = 0; l < NLAYER; ++l) w2p[l]  = (half_t*)carve(8 * 4 * 512 * 2);
  for (int l = 0; l < NLAYER; ++l) cf1p[l] = (half_t*)carve(8 * 4 * 512 * 2);
  for (int l = 0; l < NLAYER; ++l) cf2p[l] = (half_t*)carve(8 * 4 * 512 * 2);
  for (int l = 0; l < NLAYER; ++l) linp[l] = (half_t*)carve(8 * 4 * 512 * 2);
  half_t* out1p = (half_t*)carve(4 * 4 * 512 * 2);

  // ---- weight packing into WMMA B-fragment layout ----
  for (int l = 0; l < NLAYER; ++l) {
    pack_b_kernel<<<(8 * 2 * 512 + 255) / 256, 256, 0, stream>>>(
        w1 + (size_t)l * 50 * HID, w1p[l], 50, HID, 2);
    pack_b_kernel<<<(8 * 4 * 512 + 255) / 256, 256, 0, stream>>>(
        w2 + (size_t)l * HID * HID, w2p[l], HID, HID, 4);
    pack_b_kernel<<<(8 * 4 * 512 + 255) / 256, 256, 0, stream>>>(
        cf1 + (size_t)l * HID * HID, cf1p[l], HID, HID, 4);
    pack_b_kernel<<<(8 * 4 * 512 + 255) / 256, 256, 0, stream>>>(
        cf2 + (size_t)l * HID * HID, cf2p[l], HID, HID, 4);
    pack_b_kernel<<<(8 * 4 * 512 + 255) / 256, 256, 0, stream>>>(
        linw + (size_t)l * HID * HID, linp[l], HID, HID, 4);
  }
  pack_b_kernel<<<(4 * 4 * 512 + 255) / 256, 256, 0, stream>>>(o1w, out1p, HID,
                                                               64, 4);

  zero_kernel<<<1, 256, 0, stream>>>(out, NBATCH);
  embed_kernel<<<(N_NODES * HID + 255) / 256, 256, 0, stream>>>(z, emb, hbuf,
                                                                h16);

  const int nodeTiles = N_NODES / 16;  // 1250
  for (int l = 0; l < NLAYER; ++l) {
    // xh = h @ cf1 (no bias)
    gemm16_kernel<<<nodeTiles, 256, 0, stream>>>(h16, cf1p[l], nullptr, xh,
                                                 nullptr, HID, 0);
    zero_kernel<<<(N_NODES * HID + 255) / 256, 256, 0, stream>>>(agg,
                                                                 N_NODES * HID);
    // fused edge-feature + 2-layer filter MLP + gather/scatter
    edge_layer_kernel<<<N_EDGES / 128, 256, SM_TOTAL, stream>>>(
        eidx, pos, eattr, w1p[l], b1 + (size_t)l * HID, w2p[l],
        b2 + (size_t)l * HID, xh, agg);
    f32to16_kernel<<<(N_NODES * HID + 255) / 256, 256, 0, stream>>>(
        agg, agg16, N_NODES * HID);
    // t = ssp(agg @ cf2 + b)
    gemm16_kernel<<<nodeTiles, 256, 0, stream>>>(agg16, cf2p[l],
                                                 cf2b + (size_t)l * HID,
                                                 nullptr, t16, HID, 1);
    // h = h + t @ lin + b   (residual; also refresh h16)
    gemm16_kernel<<<nodeTiles, 256, 0, stream>>>(t16, linp[l],
                                                 linb + (size_t)l * HID, hbuf,
                                                 h16, HID, 2);
  }
  // o1 = ssp(h @ out1 + b)  [N,64]
  gemm16_kernel<<<nodeTiles, 256, 0, stream>>>(h16, out1p, o1b, nullptr, o1buf,
                                               64, 1);
  readout_kernel<<<(N_NODES + 255) / 256, 256, 0, stream>>>(o1buf, o2w, o2b,
                                                            batch, out);
}
